// SparseDenseMatmulLayer_56684978372609
// MI455X (gfx1250) — matrix operand order repaired and checked by reading, not compiled, not run on hardware
//
#include <hip/hip_runtime.h>
#include <stdint.h>

typedef __attribute__((ext_vector_type(16))) _Float16 v16h;
typedef __attribute__((ext_vector_type(8)))  float    v8f;
typedef __attribute__((ext_vector_type(8)))  float    f32x8;

#define DENSE   2048
#define BATCH   64
#define MAXS    1024
#define KSPLIT  8
#define KCHUNK  (DENSE / KSPLIT)      // 256 k-rows per K chunk
#define KITER   (KCHUNK / 32)         // 8 WMMA k-steps per wave
#define JTILES  (DENSE / 16)          // 128 output row tiles
#define WAVES_PER_BLOCK 8

// ---------------------------------------------------------------------------
// Zero the histogram and the output (d_out / d_ws are poisoned by the harness)
// ---------------------------------------------------------------------------
__global__ void sdml_zero(int* __restrict__ hist, float* __restrict__ out) {
    int i = blockIdx.x * blockDim.x + threadIdx.x;
    if (i < DENSE * BATCH) { hist[i] = 0; out[i] = 0.0f; }
}

// ---------------------------------------------------------------------------
// Mt_int[c][b] = multiplicity of column c among batch b's active spikes.
// ---------------------------------------------------------------------------
__global__ void sdml_hist(const int* __restrict__ spike_ids,
                          const int* __restrict__ num_spikes,
                          int* __restrict__ hist) {
    int b = blockIdx.x;
    int i = threadIdx.x;
    if (i < num_spikes[b]) {
        int c = spike_ids[b * MAXS + i];
        atomicAdd(&hist[c * BATCH + b], 1);
    }
}

// ---------------------------------------------------------------------------
// Convert int counts -> f16 (counts <= 1024 are exact in f16)
// ---------------------------------------------------------------------------
__global__ void sdml_cvt(const int* __restrict__ hist, _Float16* __restrict__ Mt) {
    int i = blockIdx.x * blockDim.x + threadIdx.x;
    if (i < DENSE * BATCH) Mt[i] = (_Float16)hist[i];
}

// ---------------------------------------------------------------------------
// out^T = W (2048x2048) x Mt (2048x64) as 16x16x32 f16 WMMAs, f32 accum.
// Block = 8 waves sharing one K chunk; the 32 KB Mt K-chunk is staged in LDS
// once per block via global_load_async_to_lds_b128 (ASYNCcnt). The K loop is
// software-pipelined: iteration kb+1's W runs (global) and B fragments (LDS)
// are fetched into fresh registers before kb's WMMAs issue, so no WMMA waits
// on a just-issued load. K split 8 ways; partials merged with f32 atomics.
// ---------------------------------------------------------------------------
__global__ void __launch_bounds__(256)
sdml_wmma_gemm(const float* __restrict__ W,
               const _Float16* __restrict__ Mt,
               float* __restrict__ out) {
    __shared__ _Float16 MtLds[KCHUNK * BATCH];   // 256 x 64 f16 = 32 KB

    int lane   = threadIdx.x & 31;
    int waveIn = threadIdx.x >> 5;

    int kChunk = blockIdx.x & (KSPLIT - 1);      // 0..7
    int jGroup = blockIdx.x >> 3;                // 0..15
    int jTile  = jGroup * WAVES_PER_BLOCK + waveIn;

    int kBase = kChunk * KCHUNK;

    // ---- Stage Mt[kBase .. kBase+256) x [0..64) into LDS (async DMA) ----
    {
        const char* src = (const char*)(Mt + (size_t)kBase * BATCH); // 32 KB
        unsigned    ldsBase = (unsigned)(size_t)(&MtLds[0]);
        // 32 KB / 16 B = 2048 chunks; 256 threads -> 8 chunks each
        for (int t = threadIdx.x; t < (KCHUNK * BATCH * 2) / 16; t += 256) {
            unsigned lofs = ldsBase + t * 16;
            uint64_t ga   = (uint64_t)src + (uint64_t)t * 16;
            asm volatile("global_load_async_to_lds_b128 %0, %1, off"
                         :: "v"(lofs), "v"(ga) : "memory");
        }
        asm volatile("s_wait_asynccnt 0" ::: "memory");
        __syncthreads();
    }

    int m = lane & 15;
    int h = lane >> 4;
    const float* Wrow = W + (size_t)(jTile * 16 + m) * DENSE;

    v8f acc0 = {}, acc1 = {}, acc2 = {}, acc3 = {};

    // ---- Pipeline prologue: operands for kb = 0 ----
    f32x8 lo = *(const f32x8*)(Wrow + kBase + h * 8);
    f32x8 hi = *(const f32x8*)(Wrow + kBase + 16 + h * 8);
    const _Float16* B0 = &MtLds[m];
    v16h b0, b1, b2, b3;
#pragma unroll
    for (int s = 0; s < 16; ++s) {
        int kk = (s < 8) ? (h * 8 + s) : (16 + h * 8 + (s - 8));
        b0[s] = B0[kk * BATCH + 0];  b1[s] = B0[kk * BATCH + 16];
        b2[s] = B0[kk * BATCH + 32]; b3[s] = B0[kk * BATCH + 48];
    }

#pragma unroll
    for (int kb = 0; kb < KITER; ++kb) {
        // Convert current A runs (packed f32->f16 cvts)
        v16h a;
#pragma unroll
        for (int e = 0; e < 8; ++e) {
            a[e]     = (_Float16)lo[e];
            a[e + 8] = (_Float16)hi[e];
        }

        // Fetch NEXT iteration's operands into fresh registers first
        f32x8 nlo = lo, nhi = hi;
        v16h  n0 = b0, n1 = b1, n2 = b2, n3 = b3;
        if (kb + 1 < KITER) {
            int k1 = kBase + (kb + 1) * 32;
            __builtin_prefetch(Wrow + k1 + 32 + h * 8, 0, 1); // 2 steps ahead
            nlo = *(const f32x8*)(Wrow + k1 + h * 8);
            nhi = *(const f32x8*)(Wrow + k1 + 16 + h * 8);
            const _Float16* Bn = &MtLds[(kb + 1) * 32 * BATCH + m];
#pragma unroll
            for (int s = 0; s < 16; ++s) {
                int kk = (s < 8) ? (h * 8 + s) : (16 + h * 8 + (s - 8));
                n0[s] = Bn[kk * BATCH + 0];  n1[s] = Bn[kk * BATCH + 16];
                n2[s] = Bn[kk * BATCH + 32]; n3[s] = Bn[kk * BATCH + 48];
            }
        }

        // WMMAs for the CURRENT iteration (operands loaded last iteration)
        acc0 = __builtin_amdgcn_wmma_f32_16x16x32_f16(false, a, false, b0, (short)0, acc0, false, false);
        acc1 = __builtin_amdgcn_wmma_f32_16x16x32_f16(false, a, false, b1, (short)0, acc1, false, false);
        acc2 = __builtin_amdgcn_wmma_f32_16x16x32_f16(false, a, false, b2, (short)0, acc2, false, false);
        acc3 = __builtin_amdgcn_wmma_f32_16x16x32_f16(false, a, false, b3, (short)0, acc3, false, false);

        lo = nlo; hi = nhi;
        b0 = n0; b1 = n1; b2 = n2; b3 = n3;
    }

    // --- Epilogue: merge K-split partials into out[b][j] with f32 atomics ---
    int jb = jTile * 16 + h * 8;
    int bn = lane & 15;
#pragma unroll
    for (int r = 0; r < 8; ++r) {
        int j = jb + r;
        unsafeAtomicAdd(&out[(size_t)(bn +  0) * DENSE + j], acc0[r]);
        unsafeAtomicAdd(&out[(size_t)(bn + 16) * DENSE + j], acc1[r]);
        unsafeAtomicAdd(&out[(size_t)(bn + 32) * DENSE + j], acc2[r]);
        unsafeAtomicAdd(&out[(size_t)(bn + 48) * DENSE + j], acc3[r]);
    }
}

// ---------------------------------------------------------------------------
extern "C" void kernel_launch(void* const* d_in, const int* in_sizes, int n_in,
                              void* d_out, int out_size, void* d_ws, size_t ws_size,
                              hipStream_t stream) {
    const int*   spike_ids  = (const int*)d_in[0];   // [64, 1024] int32
    const int*   num_spikes = (const int*)d_in[1];   // [64] int32
    const float* W          = (const float*)d_in[2]; // [2048, 2048] f32
    float*       out        = (float*)d_out;         // [64, 2048] f32

    int*      hist = (int*)d_ws;                                             // 512 KB
    _Float16* Mt   = (_Float16*)((char*)d_ws + DENSE * BATCH * sizeof(int)); // 256 KB

    int n = DENSE * BATCH;
    sdml_zero<<<(n + 255) / 256, 256, 0, stream>>>(hist, out);
    sdml_hist<<<BATCH, MAXS, 0, stream>>>(spike_ids, num_spikes, hist);
    sdml_cvt <<<(n + 255) / 256, 256, 0, stream>>>(hist, Mt);

    // 16 j-groups x 8 K-chunks = 128 blocks of 8 waves (1024 waves total)
    sdml_wmma_gemm<<<JTILES / WAVES_PER_BLOCK * KSPLIT, 256, 0, stream>>>(W, Mt, out);
}
